// PatchedQwen2MoeSparseMoeBlock_15341623181598
// MI455X (gfx1250) — matrix operand order repaired and checked by reading, not compiled, not run on hardware
//
#include <hip/hip_runtime.h>
#include <hip/hip_bf16.h>

// ---------------------------------------------------------------------------
// Qwen2-MoE sparse MoE block for MI455X (gfx1250, wave32, WMMA).
//   B=1, S=T=1024, H=2048, E=32, F=1408, SF=5632, top_k=4
// Memory-bound: ~1.25 GB unique fp32 weights @ 23.3 TB/s ~= 54 us floor.
//   K1  router: logits -> softmax -> top4 (wave32 == 32 experts, shfl reduce)
//   K1b/K1c    : counting-sort token gather (8x less expert work than dense)
//   K2  fused expert MLP per (expert, 32-token tile):
//         x-tile f16 in LDS -> gate/up GEMM (fused: 2 A + 2 B loads -> 4 WMMA)
//         -> SiLU*up act f16 in LDS -> down GEMM (1 B load -> 2 WMMA)
//         -> weighted global_atomic_add_f32.  act never touches HBM.
//   K3  shared expert gate/up -> f16 act in ws (sigmoid token-gate folded in)
//   K4  shared down proj -> atomic add into out
// 32-token M-tiles halve weight streaming vs 16; A and B fragments use the same
// contiguous k order (k = 16*(lane/16) + s).
// ---------------------------------------------------------------------------

typedef __attribute__((ext_vector_type(16))) _Float16 v16h;
typedef __attribute__((ext_vector_type(8)))  float    v8f;

#define T_TOK  1024
#define H_DIM  2048
#define N_EXP  32
#define F_DIM  1408
#define SF_DIM 5632
#define TOPK   4
#define M_TILE 32

#define LDA_X (H_DIM + 8)   // padded f16 LDS row for x tile
#define LDA_A (F_DIM + 8)   // padded f16 LDS row for act tile

// workspace layout (bytes)
#define WS_TOPK_IDX 0
#define WS_TOPK_W   16384
#define WS_SGV      32768
#define WS_COUNTS   36864
#define WS_OFFSETS  36992
#define WS_CURSOR   37120
#define WS_PAIR_TOK 37248
#define WS_PAIR_W   53632
#define WS_SACT     70144   // _Float16[T_TOK * SF_DIM]  (~11.5 MB)

// ---------------- WMMA fragment loaders (contiguous k order) ----------------

// A matrix 16x32 f16.  lane<16: M=lane, K = k0..k0+15
//                      lane>=16: M=lane-16, K = k0+16..k0+31
__device__ inline v16h load_a16(const _Float16* __restrict__ base, int lda,
                                int k0, int lane) {
  const _Float16* p = base + (lane & 15) * lda + k0 + ((lane >> 4) << 4);
  v16h a;
#pragma unroll
  for (int i = 0; i < 8; ++i) a[i] = p[i];
#pragma unroll
  for (int i = 0; i < 8; ++i) a[8 + i] = p[8 + i];
  return a;
}

// B matrix 32x16: lane n (0..15) holds weight-row n, K = k0..k0+15;
// lanes 16..31 hold rows again with K = k0+16..k0+31.  fp32 row-major
// [out_feature, K] weights, converted to f16 in registers.
__device__ inline v16h load_b32w(const float* __restrict__ base, long long ldk,
                                 int k0, int lane) {
  const float* p = base + (long long)(lane & 15) * ldk + k0 + ((lane >> 4) << 4);
  const float4* q = (const float4*)p;
  v16h b;
#pragma unroll
  for (int i = 0; i < 4; ++i) {
    float4 v = q[i];
    b[4 * i + 0] = (_Float16)v.x;
    b[4 * i + 1] = (_Float16)v.y;
    b[4 * i + 2] = (_Float16)v.z;
    b[4 * i + 3] = (_Float16)v.w;
  }
  return b;
}

#define WMMA(a, b, c) \
  __builtin_amdgcn_wmma_f32_16x16x32_f16(false, (a), false, (b), (short)0, (c), false, false)

// 32xK A (two row-tiles) x two 16-row B blocks: 4 WMMAs per {a0,a1,b0,b1}.
__device__ inline void gemm_2x2(const _Float16* __restrict__ A, int lda,
                                const float* __restrict__ B0,
                                const float* __restrict__ B1, long long ldk,
                                int K, int lane, v8f& c00, v8f& c01, v8f& c10,
                                v8f& c11) {
  for (int k0 = 0; k0 < K; k0 += 32) {
    v16h a0 = load_a16(A, lda, k0, lane);
    v16h a1 = load_a16(A + 16 * lda, lda, k0, lane);
    v16h b0 = load_b32w(B0, ldk, k0, lane);
    v16h b1 = load_b32w(B1, ldk, k0, lane);
    c00 = WMMA(a0, b0, c00);
    c01 = WMMA(a0, b1, c01);
    c10 = WMMA(a1, b0, c10);
    c11 = WMMA(a1, b1, c11);
  }
}

// 32xK A (two row-tiles) x one 16-row B block: 2 WMMAs per B load.
__device__ inline void gemm_2x1(const _Float16* __restrict__ A, long long lda,
                                const float* __restrict__ B, long long ldk,
                                int K, int lane, v8f& c0, v8f& c1) {
  for (int k0 = 0; k0 < K; k0 += 32) {
    v16h a0 = load_a16(A, (int)lda, k0, lane);
    v16h a1 = load_a16(A + 16 * lda, (int)lda, k0, lane);
    v16h b = load_b32w(B, ldk, k0, lane);
    c0 = WMMA(a0, b, c0);
    c1 = WMMA(a1, b, c1);
  }
}

__device__ inline float silu(float v) { return v / (1.f + expf(-v)); }

// ------------------------------- K1: router ---------------------------------

__global__ void router_kernel(const float* __restrict__ x,
                              const float* __restrict__ gw,
                              const float* __restrict__ segw,
                              float* __restrict__ rlog,
                              int* __restrict__ tidx, float* __restrict__ twts,
                              float* __restrict__ sgv, int* __restrict__ counts) {
  __shared__ float xrow[H_DIM];
  __shared__ float part[256];
  __shared__ float sp[256];
  int t = blockIdx.x, tid = threadIdx.x;
  const float* xr = x + (size_t)t * H_DIM;
  for (int k = tid; k < H_DIM; k += 256) xrow[k] = xr[k];
  __syncthreads();

  int e = tid & 31, seg = tid >> 5;
  const float* gr = gw + (size_t)e * H_DIM + seg * 256;
  float s = 0.f;
  for (int k = 0; k < 256; ++k) s += xrow[seg * 256 + k] * gr[k];
  part[tid] = s;
  float s2 = 0.f;
  for (int k = tid * 8; k < tid * 8 + 8; ++k) s2 += xrow[k] * segw[k];
  sp[tid] = s2;
  __syncthreads();
  for (int st = 128; st > 0; st >>= 1) {
    if (tid < st) sp[tid] += sp[tid + st];
    __syncthreads();
  }
  if (tid == 0) sgv[t] = 1.f / (1.f + expf(-sp[0]));

  if (tid < 32) {  // wave 0: lane == expert (wave32!)
    float logit = 0.f;
    for (int p = 0; p < 8; ++p) logit += part[p * 32 + tid];
    rlog[t * N_EXP + tid] = logit;
    float mx = logit;
    for (int off = 16; off >= 1; off >>= 1)
      mx = fmaxf(mx, __shfl_xor(mx, off, 32));
    float ex = expf(logit - mx);
    float sm = ex;
    for (int off = 16; off >= 1; off >>= 1) sm += __shfl_xor(sm, off, 32);
    float myp = ex / sm;
    for (int r = 0; r < TOPK; ++r) {  // deterministic argmax (low index wins)
      float bp = myp;
      int be = tid;
      for (int off = 16; off >= 1; off >>= 1) {
        float op = __shfl_xor(bp, off, 32);
        int oe = __shfl_xor(be, off, 32);
        if (op > bp || (op == bp && oe < be)) { bp = op; be = oe; }
      }
      if (tid == be) myp = -1.f;
      if (tid == r) {
        tidx[t * TOPK + r] = be;
        twts[t * TOPK + r] = bp;
        atomicAdd(&counts[be], 1);
      }
    }
  }
}

// ------------------- K1b/K1c: counting-sort token gather --------------------

__global__ void scan_kernel(const int* __restrict__ counts,
                            int* __restrict__ offsets, int* __restrict__ cursor) {
  if (threadIdx.x == 0) {
    int run = 0;
    for (int e = 0; e < N_EXP; ++e) {
      offsets[e] = run;
      cursor[e] = run;
      run += counts[e];
    }
  }
}

__global__ void scatter_kernel(const int* __restrict__ tidx,
                               const float* __restrict__ twts,
                               int* __restrict__ cursor, int* __restrict__ ptok,
                               float* __restrict__ pw) {
  int i = blockIdx.x * blockDim.x + threadIdx.x;
  if (i >= T_TOK * TOPK) return;
  int e = tidx[i];
  int pos = atomicAdd(&cursor[e], 1);
  ptok[pos] = i >> 2;
  pw[pos] = twts[i];
}

// ------------- K2: fused sparse expert MLP (expert, 32-token tile) ----------

__global__ void moe_expert_kernel(const float* __restrict__ x,
                                  const float* __restrict__ gup,
                                  const float* __restrict__ dw,
                                  const int* __restrict__ counts,
                                  const int* __restrict__ offsets,
                                  const int* __restrict__ ptok,
                                  const float* __restrict__ pw,
                                  float* __restrict__ out) {
  int e = blockIdx.y;
  int cnt = counts[e];
  int tile0 = blockIdx.x * M_TILE;
  if (tile0 >= cnt) return;  // uniform early-out

  extern __shared__ _Float16 smem[];
  _Float16* xs = smem;                     // [32][LDA_X] f16  (~128.5 KB)
  _Float16* acts = smem + M_TILE * LDA_X;  // [32][LDA_A] f16  (~88.5 KB)
  __shared__ int tok[M_TILE];
  __shared__ float twt[M_TILE];

  int tid = threadIdx.x, lane = tid & 31, w = tid >> 5;
  if (tid < M_TILE) {
    int s = tile0 + tid;
    if (s < cnt) {
      tok[tid] = ptok[offsets[e] + s];
      twt[tid] = pw[offsets[e] + s];
    } else {
      tok[tid] = 0;
      twt[tid] = 0.f;
    }
  }
  __syncthreads();

  // stage gathered x rows, fp32 -> f16 (8 threads per row, float4 loads)
  {
    int row = tid >> 3, col8 = tid & 7;
    const float4* src = (const float4*)(x + (size_t)tok[row] * H_DIM);
    _Float16* dst = xs + row * LDA_X;
    for (int c = col8; c < H_DIM / 4; c += 8) {
      float4 v = src[c];
      dst[4 * c + 0] = (_Float16)v.x;
      dst[4 * c + 1] = (_Float16)v.y;
      dst[4 * c + 2] = (_Float16)v.z;
      dst[4 * c + 3] = (_Float16)v.w;
    }
  }
  __syncthreads();

  // phase 1: gu = x @ W_gu^T ; act = silu(gate) * up  -> LDS
  const float* gbase = gup + (long long)e * (2 * F_DIM) * H_DIM;
  for (int ft = w; ft < F_DIM / 16; ft += 8) {
    v8f g0 = {}, u0 = {}, g1 = {}, u1 = {};
    gemm_2x2(xs, LDA_X, gbase + (long long)(ft * 16) * H_DIM,
             gbase + (long long)(F_DIM + ft * 16) * H_DIM, H_DIM, H_DIM, lane,
             g0, u0, g1, u1);
    int n = lane & 15;
#pragma unroll
    for (int i = 0; i < 8; ++i) {
      int m = i + ((lane >> 4) << 3);
      acts[m * LDA_A + ft * 16 + n] = (_Float16)(silu(g0[i]) * u0[i]);
      acts[(m + 16) * LDA_A + ft * 16 + n] = (_Float16)(silu(g1[i]) * u1[i]);
    }
  }
  __syncthreads();

  // phase 2: out += combine_w * (act @ down^T)
  const float* dbase = dw + (long long)e * H_DIM * F_DIM;
  for (int ht = w; ht < H_DIM / 16; ht += 8) {
    v8f c0 = {}, c1 = {};
    gemm_2x1(acts, LDA_A, dbase + (long long)(ht * 16) * F_DIM, F_DIM, F_DIM,
             lane, c0, c1);
    int n = lane & 15;
#pragma unroll
    for (int i = 0; i < 8; ++i) {
      int m = i + ((lane >> 4) << 3);
      float w0 = twt[m], w1 = twt[m + 16];
      if (w0 != 0.f)
        atomicAdd(&out[(size_t)tok[m] * H_DIM + ht * 16 + n], w0 * c0[i]);
      if (w1 != 0.f)
        atomicAdd(&out[(size_t)tok[m + 16] * H_DIM + ht * 16 + n], w1 * c1[i]);
    }
  }
}

// --------------- K3: shared expert gate/up -> act (32-token tile) -----------

__global__ void shared_act_kernel(const float* __restrict__ x,
                                  const float* __restrict__ sgw,
                                  const float* __restrict__ suw,
                                  const float* __restrict__ sgv,
                                  _Float16* __restrict__ sact) {
  extern __shared__ _Float16 smem[];
  _Float16* xs = smem;  // [32][LDA_X]
  __shared__ float sg[M_TILE];
  int tt = blockIdx.x, tid = threadIdx.x, lane = tid & 31, w = tid >> 5;
  int t0 = tt * M_TILE;
  if (tid < M_TILE) sg[tid] = sgv[t0 + tid];
  {
    int row = tid >> 3, col8 = tid & 7;
    const float4* src = (const float4*)(x + (size_t)(t0 + row) * H_DIM);
    _Float16* dst = xs + row * LDA_X;
    for (int c = col8; c < H_DIM / 4; c += 8) {
      float4 v = src[c];
      dst[4 * c + 0] = (_Float16)v.x;
      dst[4 * c + 1] = (_Float16)v.y;
      dst[4 * c + 2] = (_Float16)v.z;
      dst[4 * c + 3] = (_Float16)v.w;
    }
  }
  __syncthreads();
  for (int ft = w; ft < SF_DIM / 16; ft += 8) {
    v8f g0 = {}, u0 = {}, g1 = {}, u1 = {};
    gemm_2x2(xs, LDA_X, sgw + (long long)(ft * 16) * H_DIM,
             suw + (long long)(ft * 16) * H_DIM, H_DIM, H_DIM, lane, g0, u0,
             g1, u1);
    int n = lane & 15;
#pragma unroll
    for (int i = 0; i < 8; ++i) {
      int m = i + ((lane >> 4) << 3);
      // fold sigmoid token gate here (linear, commutes with down proj)
      sact[(size_t)(t0 + m) * SF_DIM + ft * 16 + n] =
          (_Float16)(silu(g0[i]) * u0[i] * sg[m]);
      sact[(size_t)(t0 + m + 16) * SF_DIM + ft * 16 + n] =
          (_Float16)(silu(g1[i]) * u1[i] * sg[m + 16]);
    }
  }
}

// ----------------------- K4: shared expert down proj ------------------------

__global__ void shared_down_kernel(const _Float16* __restrict__ sact,
                                   const float* __restrict__ sdw,
                                   float* __restrict__ out) {
  int tt = blockIdx.x, tid = threadIdx.x, lane = tid & 31, w = tid >> 5;
  int ht = blockIdx.y * 8 + w;  // 128 h-tiles total
  int t0 = tt * M_TILE;
  const _Float16* A = sact + (size_t)t0 * SF_DIM;
  const float* B = sdw + (long long)(ht * 16) * SF_DIM;
  v8f c0 = {}, c1 = {};
  gemm_2x1(A, SF_DIM, B, SF_DIM, SF_DIM, lane, c0, c1);
  int n = lane & 15;
#pragma unroll
  for (int i = 0; i < 8; ++i) {
    int m = i + ((lane >> 4) << 3);
    atomicAdd(&out[(size_t)(t0 + m) * H_DIM + ht * 16 + n], c0[i]);
    atomicAdd(&out[(size_t)(t0 + m + 16) * H_DIM + ht * 16 + n], c1[i]);
  }
}

// --------------------------------- launch -----------------------------------

extern "C" void kernel_launch(void* const* d_in, const int* in_sizes, int n_in,
                              void* d_out, int out_size, void* d_ws,
                              size_t ws_size, hipStream_t stream) {
  const float* x    = (const float*)d_in[0];
  const float* gw   = (const float*)d_in[1];
  const float* gup  = (const float*)d_in[2];
  const float* dw   = (const float*)d_in[3];
  const float* sgw  = (const float*)d_in[4];
  const float* suw  = (const float*)d_in[5];
  const float* sdw  = (const float*)d_in[6];
  const float* segw = (const float*)d_in[7];
  // d_in[8] = top_k (hardcoded 4)
  (void)in_sizes; (void)n_in; (void)out_size; (void)ws_size;

  float* out  = (float*)d_out;
  float* rlog = out + (size_t)T_TOK * H_DIM;

  char* ws = (char*)d_ws;
  int*      topk_idx = (int*)(ws + WS_TOPK_IDX);
  float*    topk_w   = (float*)(ws + WS_TOPK_W);
  float*    sgv      = (float*)(ws + WS_SGV);
  int*      counts   = (int*)(ws + WS_COUNTS);
  int*      offsets  = (int*)(ws + WS_OFFSETS);
  int*      cursor   = (int*)(ws + WS_CURSOR);
  int*      ptok     = (int*)(ws + WS_PAIR_TOK);
  float*    pwts     = (float*)(ws + WS_PAIR_W);
  _Float16* sact     = (_Float16*)(ws + WS_SACT);

  hipMemsetAsync(out, 0, (size_t)T_TOK * H_DIM * sizeof(float), stream);
  hipMemsetAsync(counts, 0, N_EXP * sizeof(int), stream);

  router_kernel<<<T_TOK, 256, 0, stream>>>(x, gw, segw, rlog, topk_idx,
                                           topk_w, sgv, counts);
  scan_kernel<<<1, 32, 0, stream>>>(counts, offsets, cursor);
  scatter_kernel<<<(T_TOK * TOPK + 255) / 256, 256, 0, stream>>>(
      topk_idx, topk_w, cursor, ptok, pwts);

  size_t smem2 = (size_t)M_TILE * (LDA_X + LDA_A) * sizeof(_Float16);  // ~217 KB
  moe_expert_kernel<<<dim3(T_TOK / M_TILE, N_EXP), 256, smem2, stream>>>(
      x, gup, dw, counts, offsets, ptok, pwts, out);

  size_t smem3 = (size_t)M_TILE * LDA_X * sizeof(_Float16);  // ~128.5 KB
  shared_act_kernel<<<T_TOK / M_TILE, 256, smem3, stream>>>(x, sgw, suw, sgv,
                                                            sact);
  shared_down_kernel<<<dim3(T_TOK / M_TILE, H_DIM / 128), 256, 0, stream>>>(
      sact, sdw, out);
}
